// GATOTFSDetector_62216896249979
// MI455X (gfx1250) — matrix operand order repaired and checked by reading, not compiled, not run on hardware
//
#include <hip/hip_runtime.h>
#include <hip/hip_bf16.h>
#include <stdint.h>

#define NB    16      // batch
#define NN    1024    // nodes
#define FF    8       // F
#define FP    16      // F'
#define TT    10      // iterations
#define NH1   64
#define NH2   32
#define SS    2
#define EPS   1e-8f
#define SLOPE 0.2f

typedef __attribute__((ext_vector_type(16))) _Float16 v16h;
typedef __attribute__((ext_vector_type(8)))  float    v8f;

union AFrag { v16h v; uint32_t u[8]; };

// ---------------------------------------------------------------------------
// Kernel 1: node features z = H^T y, g = sum_i H_ij^2, then 3->8->8 init MLP
// ---------------------------------------------------------------------------
__global__ void k_node_init(const float* __restrict__ H, const float* __restrict__ y,
                            const float* __restrict__ sigma2,
                            const float* __restrict__ w1, const float* __restrict__ b1,
                            const float* __restrict__ w2, const float* __restrict__ b2,
                            float* __restrict__ u0) {
  int t = blockIdx.x * blockDim.x + threadIdx.x;       // t = b*NN + j
  if (t >= NB * NN) return;
  int b = t >> 10, j = t & (NN - 1);
  const float* Hb = H + (size_t)b * NN * NN;
  const float* yb = y + (size_t)b * NN;
  float z = 0.f, g = 0.f;
  for (int i = 0; i < NN; ++i) {
    float v = Hb[(size_t)i * NN + j];                  // coalesced across j
    z = fmaf(v, yb[i], z);
    g = fmaf(v, v, g);
  }
  float feats[3] = { z, g, sigma2[b] };
  float t1[FF];
  #pragma unroll
  for (int k = 0; k < FF; ++k) {
    float acc = b1[k];
    #pragma unroll
    for (int d = 0; d < 3; ++d) acc = fmaf(feats[d], w1[d * FF + k], acc);
    t1[k] = acc > 0.f ? acc : 0.f;
  }
  #pragma unroll
  for (int f = 0; f < FF; ++f) {
    float acc = b2[f];
    #pragma unroll
    for (int k = 0; k < FF; ++k) acc = fmaf(t1[k], w2[k * FF + f], acc);
    u0[(size_t)t * FF + f] = acc;
  }
}

// ---------------------------------------------------------------------------
// Kernel 2: masked edge logits e[b,i,j] = a0*H_ij + a1*H_ji  (f16),
//           -inf where no edge (mask folded in as sentinel). LDS-tiled transpose.
// ---------------------------------------------------------------------------
__global__ void k_edge_logits(const float* __restrict__ H, const float* __restrict__ a_edge,
                              _Float16* __restrict__ e) {
  __shared__ float T1[32][33];
  __shared__ float T2[32][33];
  int tilesPerRow = NN / 32;                           // 32
  int bid = blockIdx.x;
  int b  = bid / (tilesPerRow * tilesPerRow);
  int rr = bid % (tilesPerRow * tilesPerRow);
  int i0 = (rr / tilesPerRow) * 32;
  int j0 = (rr % tilesPerRow) * 32;
  const float* Hb = H + (size_t)b * NN * NN;
  float a0 = a_edge[0], a1 = a_edge[1];
  for (int k = threadIdx.x; k < 1024; k += blockDim.x) {
    int r = k >> 5, c = k & 31;
    T1[r][c] = Hb[(size_t)(i0 + r) * NN + (j0 + c)];
    T2[r][c] = Hb[(size_t)(j0 + r) * NN + (i0 + c)];
  }
  __syncthreads();
  for (int k = threadIdx.x; k < 1024; k += blockDim.x) {
    int r = k >> 5, c = k & 31;
    float hij = T1[r][c];
    float hji = T2[c][r];
    bool msk = (fabsf(hij) > EPS) || (fabsf(hji) > EPS) || ((i0 + r) == (j0 + c));
    float ev = a0 * hij + a1 * hji;
    e[((size_t)b * NN + (i0 + r)) * NN + (j0 + c)] =
        msk ? (_Float16)ev : (_Float16)(-__builtin_inff());
  }
}

// ---------------------------------------------------------------------------
// Kernel 3 (per iter): h = u@W (f32), write hT in f16 [b][f'][j]; s_src, s_dst
// ---------------------------------------------------------------------------
__global__ void k_prep(const float* __restrict__ u, const float* __restrict__ W,
                       const float* __restrict__ a_src, const float* __restrict__ a_dst,
                       _Float16* __restrict__ hT, float* __restrict__ ssrc,
                       float* __restrict__ sdst) {
  int t = blockIdx.x * blockDim.x + threadIdx.x;
  if (t >= NB * NN) return;
  int b = t >> 10, j = t & (NN - 1);
  float uu[FF];
  #pragma unroll
  for (int f = 0; f < FF; ++f) uu[f] = u[(size_t)t * FF + f];
  float ss = 0.f, dd = 0.f;
  #pragma unroll
  for (int fp = 0; fp < FP; ++fp) {
    float acc = 0.f;
    #pragma unroll
    for (int f = 0; f < FF; ++f) acc = fmaf(uu[f], W[f * FP + fp], acc);
    hT[((size_t)b * FP + fp) * NN + j] = (_Float16)acc;
    ss = fmaf(acc, a_src[fp], ss);
    dd = fmaf(acc, a_dst[fp], dd);
  }
  ssrc[t] = ss;
  sdst[t] = dd;
}

// ---------------------------------------------------------------------------
// Kernel 4 (per iter): GAT attention + aggregation via v_wmma_f32_16x16x32_f16
// Block = 128 threads (4 waves); each wave owns a 16-row tile. Two-pass softmax:
// pass1 row max, pass2 exp into f16 A-tiles fed to WMMA; divide accum by row sum.
// ---------------------------------------------------------------------------
#define HSTR 1034          // f16 stride for hT in LDS (bank-conflict pad)
#define ASTR 36            // f16 stride for score tile rows

__global__ void k_attn(const _Float16* __restrict__ e, const _Float16* __restrict__ hTg,
                       const float* __restrict__ ssrc, const float* __restrict__ sdst,
                       float* __restrict__ mOut) {
  __shared__ _Float16 shT[FP * HSTR];
  __shared__ float    sd[NN];
  __shared__ _Float16 sTile[4][16 * ASTR];

  int tid  = threadIdx.x;
  int wave = tid >> 5;
  int lane = tid & 31;
  int blocksPerB = NN / 64;                            // 16
  int b       = blockIdx.x / blocksPerB;
  int rowBase = (blockIdx.x % blocksPerB) * 64 + wave * 16;

  // cooperative load hT[b] (f16, 16x1024) into LDS with padded stride
  {
    const uint32_t* src = (const uint32_t*)(hTg + (size_t)b * FP * NN);
    uint32_t* dst = (uint32_t*)shT;
    for (int w = tid; w < FP * (NN / 2); w += 128) {
      int f = w >> 9, jw = w & 511;
      dst[f * (HSTR / 2) + jw] = src[w];
    }
    for (int idx = tid; idx < NN; idx += 128) sd[idx] = sdst[(size_t)b * NN + idx];
  }
  __syncthreads();

  float sr[16];
  #pragma unroll
  for (int r = 0; r < 16; ++r) sr[r] = ssrc[(size_t)b * NN + rowBase + r];

  const _Float16* ep = e + ((size_t)b * NN + rowBase) * NN;

  // ---- pass 1: per-row max ----
  float mx[16];
  #pragma unroll
  for (int r = 0; r < 16; ++r) mx[r] = -__builtin_inff();
  for (int jt = 0; jt < NN; jt += 32) {
    int j = jt + lane;
    float d = sd[j];
    #pragma unroll
    for (int r = 0; r < 16; ++r) {
      float x = sr[r] + d + (float)ep[(size_t)r * NN + j];
      x = x > 0.f ? x : SLOPE * x;
      mx[r] = fmaxf(mx[r], x);
    }
  }
  #pragma unroll
  for (int r = 0; r < 16; ++r) {
    float v = mx[r];
    #pragma unroll
    for (int off = 16; off > 0; off >>= 1) v = fmaxf(v, __shfl_xor(v, off, 32));
    mx[r] = v;
  }

  // ---- pass 2: unnormalized exp -> WMMA accumulate; track row sums ----
  float sm[16];
  #pragma unroll
  for (int r = 0; r < 16; ++r) sm[r] = 0.f;
  v8f c = {};

  int halfA = (lane >> 4) * 8;       // A: K offset for upper lanes
  int rowA  = lane & 15;             // A: M per lane
  int fcol  = lane & 15;             // B/D: N per lane
  int halfB = (lane >> 4) * 16;      // unused; B loads use halfA (same K order as A)
  _Float16* myTile = sTile[wave];

  for (int jt = 0; jt < NN; jt += 32) {
    int j = jt + lane;
    float d = sd[j];
    #pragma unroll
    for (int r = 0; r < 16; ++r) {
      float x = sr[r] + d + (float)ep[(size_t)r * NN + j];
      x = x > 0.f ? x : SLOPE * x;
      float p = __expf(x - mx[r]);
      sm[r] += p;
      myTile[r * ASTR + lane] = (_Float16)p;
    }
    asm volatile("s_wait_dscnt 0x0" ::: "memory");     // wave-local LDS RAW fence

    AFrag a, bm;
    #pragma unroll
    for (int p = 0; p < 8; ++p) {
      int K = ((p < 4) ? 2 * p : 16 + 2 * (p - 4)) + halfA;
      a.u[p] = *(const uint32_t*)&myTile[rowA * ASTR + K];
    }
    #pragma unroll
    for (int p = 0; p < 8; ++p) {
      int K = ((p < 4) ? 2 * p : 16 + 2 * (p - 4)) + halfA;
      bm.u[p] = *(const uint32_t*)&shT[fcol * HSTR + jt + K];
    }
    c = __builtin_amdgcn_wmma_f32_16x16x32_f16(false, a.v, false, bm.v,
                                               (short)0, c, false, false);
  }

  // ---- row-sum reduce + normalize + store m ----
  #pragma unroll
  for (int r = 0; r < 16; ++r) {
    float v = sm[r];
    #pragma unroll
    for (int off = 16; off > 0; off >>= 1) v += __shfl_xor(v, off, 32);
    sm[r] = v;
  }
  bool lo = lane < 16;
  #pragma unroll
  for (int r = 0; r < 8; ++r) {
    float s   = lo ? sm[r] : sm[r + 8];
    int rowG  = rowBase + r + (lo ? 0 : 8);
    mOut[((size_t)b * NN + rowG) * FP + fcol] = c[r] * (1.0f / s);
  }
}

// ---------------------------------------------------------------------------
// Kernel 5 (per iter): node update MLP 25->64->32->8, weights cached in LDS
// ---------------------------------------------------------------------------
__global__ void k_update(const float* __restrict__ uIn, const float* __restrict__ mBuf,
                         const float* __restrict__ sigma2,
                         const float* __restrict__ wu1, const float* __restrict__ bu1,
                         const float* __restrict__ wu2, const float* __restrict__ bu2,
                         const float* __restrict__ wu3, const float* __restrict__ bu3,
                         float* __restrict__ uOut) {
  const int DU = FF + FP + 1;                          // 25
  __shared__ float s1[DU * NH1];
  __shared__ float s2[NH1 * NH2];
  __shared__ float s3[NH2 * FF];
  __shared__ float sb1[NH1], sb2[NH2], sb3[FF];
  for (int k = threadIdx.x; k < DU * NH1;  k += blockDim.x) s1[k] = wu1[k];
  for (int k = threadIdx.x; k < NH1 * NH2; k += blockDim.x) s2[k] = wu2[k];
  for (int k = threadIdx.x; k < NH2 * FF;  k += blockDim.x) s3[k] = wu3[k];
  for (int k = threadIdx.x; k < NH1; k += blockDim.x) sb1[k] = bu1[k];
  for (int k = threadIdx.x; k < NH2; k += blockDim.x) sb2[k] = bu2[k];
  for (int k = threadIdx.x; k < FF;  k += blockDim.x) sb3[k] = bu3[k];
  __syncthreads();

  int t = blockIdx.x * blockDim.x + threadIdx.x;
  if (t >= NB * NN) return;
  int b = t >> 10;

  float cat[DU];
  #pragma unroll
  for (int f = 0; f < FF; ++f) cat[f] = uIn[(size_t)t * FF + f];
  #pragma unroll
  for (int f = 0; f < FP; ++f) cat[FF + f] = mBuf[(size_t)t * FP + f];
  cat[FF + FP] = sigma2[b];

  float x1[NH1];
  #pragma unroll 4
  for (int k = 0; k < NH1; ++k) {
    float acc = sb1[k];
    #pragma unroll
    for (int d = 0; d < DU; ++d) acc = fmaf(cat[d], s1[d * NH1 + k], acc);
    x1[k] = acc > 0.f ? acc : 0.f;
  }
  float x2[NH2];
  #pragma unroll 4
  for (int k = 0; k < NH2; ++k) {
    float acc = sb2[k];
    #pragma unroll 8
    for (int d = 0; d < NH1; ++d) acc = fmaf(x1[d], s2[d * NH2 + k], acc);
    x2[k] = acc > 0.f ? acc : 0.f;
  }
  #pragma unroll
  for (int f = 0; f < FF; ++f) {
    float acc = sb3[f];
    #pragma unroll 8
    for (int d = 0; d < NH2; ++d) acc = fmaf(x2[d], s3[d * FF + f], acc);
    uOut[(size_t)t * FF + f] = acc;
  }
}

// ---------------------------------------------------------------------------
// Kernel 6: readout logits = u @ wr + br
// ---------------------------------------------------------------------------
__global__ void k_readout(const float* __restrict__ u, const float* __restrict__ wr,
                          const float* __restrict__ br, float* __restrict__ out) {
  int t = blockIdx.x * blockDim.x + threadIdx.x;
  if (t >= NB * NN) return;
  float uu[FF];
  #pragma unroll
  for (int f = 0; f < FF; ++f) uu[f] = u[(size_t)t * FF + f];
  #pragma unroll
  for (int s = 0; s < SS; ++s) {
    float acc = br[s];
    #pragma unroll
    for (int f = 0; f < FF; ++f) acc = fmaf(uu[f], wr[f * SS + s], acc);
    out[(size_t)t * SS + s] = acc;
  }
}

// ---------------------------------------------------------------------------
extern "C" void kernel_launch(void* const* d_in, const int* in_sizes, int n_in,
                              void* d_out, int out_size, void* d_ws, size_t ws_size,
                              hipStream_t stream) {
  const float* y      = (const float*)d_in[0];
  const float* H      = (const float*)d_in[1];
  const float* sigma2 = (const float*)d_in[2];
  const float* w1     = (const float*)d_in[3];
  const float* b1     = (const float*)d_in[4];
  const float* w2     = (const float*)d_in[5];
  const float* b2     = (const float*)d_in[6];
  const float* W      = (const float*)d_in[7];
  const float* a_src  = (const float*)d_in[8];
  const float* a_dst  = (const float*)d_in[9];
  const float* a_edge = (const float*)d_in[10];
  const float* wu1    = (const float*)d_in[11];
  const float* bu1    = (const float*)d_in[12];
  const float* wu2    = (const float*)d_in[13];
  const float* bu2    = (const float*)d_in[14];
  const float* wu3    = (const float*)d_in[15];
  const float* bu3    = (const float*)d_in[16];
  const float* wr     = (const float*)d_in[17];
  const float* br     = (const float*)d_in[18];
  float* out = (float*)d_out;

  // workspace carve-up (256B aligned)
  char* ws = (char*)d_ws;
  size_t off = 0;
  auto alloc = [&](size_t bytes) { char* p = ws + off; off = (off + bytes + 255) & ~(size_t)255; return p; };
  _Float16* eM   = (_Float16*)alloc((size_t)NB * NN * NN * sizeof(_Float16)); // 32 MB
  _Float16* hT   = (_Float16*)alloc((size_t)NB * FP * NN * sizeof(_Float16));
  float*    uA   = (float*)   alloc((size_t)NB * NN * FF * sizeof(float));
  float*    uB   = (float*)   alloc((size_t)NB * NN * FF * sizeof(float));
  float*    ssrc = (float*)   alloc((size_t)NB * NN * sizeof(float));
  float*    sdst = (float*)   alloc((size_t)NB * NN * sizeof(float));
  float*    mBuf = (float*)   alloc((size_t)NB * NN * FP * sizeof(float));
  (void)ws_size;

  const int nodes = NB * NN;                                    // 16384
  k_node_init<<<nodes / 256, 256, 0, stream>>>(H, y, sigma2, w1, b1, w2, b2, uA);
  k_edge_logits<<<NB * (NN / 32) * (NN / 32), 256, 0, stream>>>(H, a_edge, eM);

  float* uCur = uA; float* uNxt = uB;
  for (int t = 0; t < TT; ++t) {
    k_prep<<<nodes / 256, 256, 0, stream>>>(uCur, W, a_src, a_dst, hT, ssrc, sdst);
    k_attn<<<NB * (NN / 64), 128, 0, stream>>>(eM, hT, ssrc, sdst, mBuf);
    k_update<<<nodes / 256, 256, 0, stream>>>(uCur, mBuf, sigma2, wu1, bu1, wu2, bu2,
                                              wu3, bu3, uNxt);
    float* tmp = uCur; uCur = uNxt; uNxt = tmp;
  }
  k_readout<<<nodes / 256, 256, 0, stream>>>(uCur, wr, br, out);
  (void)in_sizes; (void)n_in; (void)out_size;
}
